// Net_37142877176591
// MI455X (gfx1250) — matrix operand order repaired and checked by reading, not compiled, not run on hardware
//
#include <hip/hip_runtime.h>
#include <hip/hip_bf16.h>
#include <math.h>

typedef __attribute__((ext_vector_type(16))) _Float16 v16h;
typedef __attribute__((ext_vector_type(8)))  _Float16 v8h;
typedef __attribute__((ext_vector_type(4)))  _Float16 v4h;
typedef __attribute__((ext_vector_type(8)))  float    v8f;

#define NPTS  4096
#define BATCH 8
#define BN    (BATCH * NPTS)
#define KNN   20

// ---------------------------------------------------------------------------
// CDNA5 WMMA fragment builders (wave32), A/B in contiguous k order (same for both).
//
// A 16x32 f16: lane L row m=L&15, kb=(L>>4)*16. Lane's 16 halves are two
// contiguous 8-half runs: k=kb..kb+7 and k=kb+8..kb+15 -> 2x ds_load_b128
// when the LDS row stride is a multiple of 8 halves (16 B).
//
// B 32x16 f16: lane L col n=L&15, kb=(L>>4)*16, frag element e <-> k=kb+e.
// We stage B pre-packed as Bpack[lane][16] (stride 24 halves) so a fragment
// is two contiguous 8-half runs -> 2x ds_load_b128.
//
// C/D 16x16 f32: lane L, VGPR r -> row m = r + (L>>4)*8, col n = L&15.
// ---------------------------------------------------------------------------
__device__ inline v16h frag_a_pack(const _Float16* lds, int ld, int lane) {
    int m  = lane & 15;
    int kb = (lane >> 4) * 16;
    const _Float16* p = lds + m * ld + kb;
    v8h lo = *(const v8h*)(p);
    v8h hi = *(const v8h*)(p + 8);
    return __builtin_shufflevector(lo, hi, 0, 1, 2, 3, 4, 5, 6, 7,
                                   8, 9, 10, 11, 12, 13, 14, 15);
}

__device__ inline v16h frag_b_pack(const _Float16* lds, int lane) {
    const _Float16* p = lds + lane * 24;
    v8h lo = *(const v8h*)(p);
    v8h hi = *(const v8h*)(p + 8);
    return __builtin_shufflevector(lo, hi, 0, 1, 2, 3, 4, 5, 6, 7,
                                   8, 9, 10, 11, 12, 13, 14, 15);
}

__device__ inline void atomic_fmax(float* addr, float val) {
    unsigned int* ua  = (unsigned int*)addr;
    unsigned int  cur = *ua;
    while (__uint_as_float(cur) < val) {
        unsigned int prev = atomicCAS(ua, cur, __float_as_uint(val));
        if (prev == cur) break;
        cur = prev;
    }
}

// ---------------------------------------------------------------------------
// Feature prep: f32 features -> f16 (K-padded) rows + squared norms
// ---------------------------------------------------------------------------
__global__ void prep0_kernel(const float* __restrict__ pos, const float* __restrict__ x,
                             _Float16* __restrict__ Xh, float* __restrict__ sqn) {
    int i = blockIdx.x * blockDim.x + threadIdx.x;
    if (i >= BN) return;
    float v[6];
#pragma unroll
    for (int k = 0; k < 3; ++k) { v[k] = pos[i * 3 + k]; v[3 + k] = x[i * 3 + k]; }
    float s = 0.f;
#pragma unroll
    for (int k = 0; k < 6; ++k) s += v[k] * v[k];
    sqn[i] = s;
    _Float16* row = Xh + (size_t)i * 32;
#pragma unroll
    for (int k = 0; k < 6; ++k) row[k] = (_Float16)v[k];
#pragma unroll
    for (int k = 6; k < 32; ++k) row[k] = (_Float16)0.f;
}

__global__ void prep_feats_kernel(const float* __restrict__ X, _Float16* __restrict__ Xh,
                                  float* __restrict__ sqn, int C, int Kpad) {
    int i = blockIdx.x * blockDim.x + threadIdx.x;
    if (i >= BN) return;
    float s = 0.f;
    for (int k = 0; k < C; ++k) {
        float v = X[(size_t)i * C + k];
        s += v * v;
        Xh[(size_t)i * Kpad + k] = (_Float16)v;
    }
    for (int k = C; k < Kpad; ++k) Xh[(size_t)i * Kpad + k] = (_Float16)0.f;
    sqn[i] = s;
}

// W [2*Cin, Cout] f32 -> Wcat f16 [Kpad, 2*Cout]: cols [0,Cout)=W_top, [Cout,2Cout)=W_bot
__global__ void prep_edge_w_kernel(const float* __restrict__ W, _Float16* __restrict__ Wcat,
                                   int Cin, int Cout, int Kpad) {
    int t = blockIdx.x * blockDim.x + threadIdx.x;
    int total = Kpad * 2 * Cout;
    if (t >= total) return;
    int k  = t / (2 * Cout);
    int c2 = t % (2 * Cout);
    float v = 0.f;
    if (k < Cin)
        v = (c2 < Cout) ? W[(size_t)k * Cout + c2] : W[(size_t)(Cin + k) * Cout + (c2 - Cout)];
    Wcat[(size_t)k * (2 * Cout) + c2] = (_Float16)v;
}

__global__ void conv_w_kernel(const float* __restrict__ W, _Float16* __restrict__ Wh, int total) {
    int t = blockIdx.x * blockDim.x + threadIdx.x;
    if (t < total) Wh[t] = (_Float16)W[t];
}

__global__ void fill_kernel(float* __restrict__ p, int total, float v) {
    int t = blockIdx.x * blockDim.x + threadIdx.x;
    if (t < total) p[t] = v;
}

// ---------------------------------------------------------------------------
// Fused kNN: one wave per 16 query rows of one batch. Templated on the number
// of 32-wide K chunks (NCH in {1,2,4}, Kpad = 32*NCH) so the inner loop is
// straight-line: stage (unrolled load-then-store) -> NCH back-to-back WMMAs
// -> dist tile -> top-20. Per-lane top-20 in registers with incremental worst
// tracking. Self-distance included (matches reference).
// ---------------------------------------------------------------------------
template <int NCH>
__global__ void __launch_bounds__(32) knn_kernel(const _Float16* __restrict__ Xh,
                                                 const float* __restrict__ sqn,
                                                 int* __restrict__ idxOut) {
    constexpr int KPAD = 32 * NCH;
    constexpr int KQ4  = KPAD >> 2;  // 8B quads per row
    constexpr int NQ   = 4 * NCH;    // quads per lane per 16-row tile
    __shared__ __align__(16) _Float16 At[16 * 136];        // [m][k], stride 136 halves
    __shared__ __align__(16) _Float16 Bt[NCH * 32 * 24];   // packed per 32-k chunk
    __shared__ __align__(16) float    dt[16 * 17];

    int lane = threadIdx.x;
    int b    = blockIdx.y;
    int i0   = blockIdx.x * 16;
    int base = b * NPTS;
    int hi   = lane >> 4;
    int lo   = lane & 15;

    // stage A tile: coalesced 8B quads along k, two-phase
    {
        v4h q[NQ];
#pragma unroll
        for (int i = 0; i < NQ; ++i) {
            int t = lane + i * 32;
            int r = t / KQ4, c = (t % KQ4) * 4;
            q[i] = *(const v4h*)(Xh + (size_t)(base + i0 + r) * KPAD + c);
        }
#pragma unroll
        for (int i = 0; i < NQ; ++i) {
            int t = lane + i * 32;
            int r = t / KQ4, c = (t % KQ4) * 4;
            *(v4h*)(At + r * 136 + c) = q[i];
        }
    }
    __syncthreads();

    v16h afr[NCH];
#pragma unroll
    for (int ch = 0; ch < NCH; ++ch) afr[ch] = frag_a_pack(At + ch * 32, 136, lane);

    float sqi[8];
#pragma unroll
    for (int r = 0; r < 8; ++r) sqi[r] = sqn[base + i0 + r + hi * 8];

    float best[KNN];
    int   bidx[KNN];
    for (int t = 0; t < KNN; ++t) { best[t] = 3.0e38f; bidx[t] = 0; }
    float wv = 3.0e38f;
    int   wp = 0;

    for (int jt = 0; jt < NPTS / 16; ++jt) {
        int j0 = jt * 16;
        __syncthreads();
        // stage neighbor tile packed in fragment order: unrolled two-phase;
        // 8B global loads along k, 8B LDS stores (4 contiguous halves stay
        // inside one 16-k group).
        {
            v4h q[NQ];
#pragma unroll
            for (int i = 0; i < NQ; ++i) {
                int t = lane + i * 32;
                int n = t / KQ4, k = (t % KQ4) * 4;
                q[i] = *(const v4h*)(Xh + (size_t)(base + j0 + n) * KPAD + k);
            }
#pragma unroll
            for (int i = 0; i < NQ; ++i) {
                int t = lane + i * 32;
                int n = t / KQ4, k = (t % KQ4) * 4;
                int ch = k >> 5, kk = k & 31;
                *(v4h*)(Bt + ch * 768 + (((kk >> 4) << 4) + n) * 24 + (kk & 15)) = q[i];
            }
        }
        __syncthreads();

        v8f acc = {};
#pragma unroll
        for (int ch = 0; ch < NCH; ++ch) {
            v16h bf = frag_b_pack(Bt + ch * 768, lane);
            acc = __builtin_amdgcn_wmma_f32_16x16x32_f16(
                false, afr[ch], false, bf, (short)0, acc, false, false);
        }

        float sj = sqn[base + j0 + lo];
#pragma unroll
        for (int r = 0; r < 8; ++r) {
            int m = r + hi * 8;
            dt[m * 17 + lo] = sqi[r] + sj - 2.0f * acc[r];
        }
        __syncthreads();

        if (lane < 16) {
            for (int n = 0; n < 16; ++n) {
                float d = dt[lane * 17 + n];
                if (d < wv) {
                    best[wp] = d;
                    bidx[wp] = j0 + n;
                    wv = best[0]; wp = 0;
                    for (int t = 1; t < KNN; ++t)
                        if (best[t] > wv) { wv = best[t]; wp = t; }
                }
            }
        }
    }

    if (lane < 16) {
        int* o = idxOut + (size_t)(base + i0 + lane) * KNN;
        for (int t = 0; t < KNN; ++t) o[t] = bidx[t];
    }
}

// ---------------------------------------------------------------------------
// f16 GEMM C[M,N] = A[M,K] * B[K,N] (+bias). 4 waves/block; each wave owns a
// 16x32 output tile (two accumulators share one A fragment). K-loop in
// 32-chunks through LDS with unrolled two-phase staging. If rowsPerBatch>0,
// rows are max-reduced into C[batch, col] (fused global max pooling).
// ---------------------------------------------------------------------------
__global__ void __launch_bounds__(128) gemm_kernel(const _Float16* __restrict__ A,
                                                   const _Float16* __restrict__ B,
                                                   const float* __restrict__ bias,
                                                   float* __restrict__ C,
                                                   int M, int N, int K, int rowsPerBatch) {
    __shared__ __align__(16) _Float16 As[64 * 40];      // [m][k], stride 40 halves
    __shared__ __align__(16) _Float16 Bs[2 * 32 * 24];  // packed, 2 col-subtiles
    int tid  = threadIdx.x;
    int lane = tid & 31, wid = tid >> 5;
    int n0 = blockIdx.x * 32;
    int m0 = blockIdx.y * 64;

    int ar = tid >> 3;          // A row base (16-row step), col quad
    int ac = (tid & 7) * 4;
    int bk = tid >> 3;          // B k-row base (16-row step), col quad
    int bq = (tid & 7) * 4;

    v8f acc0 = {};
    v8f acc1 = {};
    for (int k0 = 0; k0 < K; k0 += 32) {
        __syncthreads();
        // two-phase staging: all global loads issued first, then LDS stores
        v4h aq[4];
#pragma unroll
        for (int i = 0; i < 4; ++i)
            aq[i] = *(const v4h*)(A + (size_t)(m0 + ar + i * 16) * K + k0 + ac);
        v4h bqv[2];
#pragma unroll
        for (int i = 0; i < 2; ++i)
            bqv[i] = *(const v4h*)(B + (size_t)(k0 + bk + i * 16) * N + n0 + bq);
#pragma unroll
        for (int i = 0; i < 4; ++i)
            *(v4h*)(As + (ar + i * 16) * 40 + ac) = aq[i];
#pragma unroll
        for (int i = 0; i < 2; ++i) {
            int k = bk + i * 16;
#pragma unroll
            for (int e = 0; e < 4; ++e) {
                int n = bq + e;
                Bs[(n >> 4) * 768 + (((k >> 4) << 4) + (n & 15)) * 24 + (k & 15)] = bqv[i][e];
            }
        }
        __syncthreads();
        v16h af  = frag_a_pack(As + (wid * 16) * 40, 40, lane);
        v16h bf0 = frag_b_pack(Bs, lane);
        v16h bf1 = frag_b_pack(Bs + 768, lane);
        acc0 = __builtin_amdgcn_wmma_f32_16x16x32_f16(
            false, af, false, bf0, (short)0, acc0, false, false);
        acc1 = __builtin_amdgcn_wmma_f32_16x16x32_f16(
            false, af, false, bf1, (short)0, acc1, false, false);
    }

    int   lo   = lane & 15;
    int   hi   = lane >> 4;
    int   col0 = n0 + lo;
    int   col1 = n0 + 16 + lo;
    float bv0  = bias ? bias[col0] : 0.f;
    float bv1  = bias ? bias[col1] : 0.f;

    if (rowsPerBatch > 0) {
        float m0v = -3.0e38f, m1v = -3.0e38f;
#pragma unroll
        for (int r = 0; r < 8; ++r) {
            m0v = fmaxf(m0v, acc0[r] + bv0);
            m1v = fmaxf(m1v, acc1[r] + bv1);
        }
        int batch = (m0 + wid * 16) / rowsPerBatch;
        atomic_fmax(&C[(size_t)batch * N + col0], m0v);
        atomic_fmax(&C[(size_t)batch * N + col1], m1v);
    } else {
#pragma unroll
        for (int r = 0; r < 8; ++r) {
            int row = m0 + wid * 16 + hi * 8 + r;
            C[(size_t)row * N + col0] = acc0[r] + bv0;
            C[(size_t)row * N + col1] = acc1[r] + bv1;
        }
    }
}

// ---------------------------------------------------------------------------
// EdgeConv epilogue: out[i,c] = u[i,c] - v[i,c] + b[c] + max_k v[nbr(i,k),c]
// where UV = [BN, 2*Cout] f32 (U = cols[0,Cout), V = cols[Cout,2Cout))
// ---------------------------------------------------------------------------
__global__ void edge_agg_kernel(const float* __restrict__ UV, const int* __restrict__ idx,
                                const float* __restrict__ bias, float* __restrict__ Xout,
                                int Cout) {
    long t     = blockIdx.x * (long)blockDim.x + threadIdx.x;
    long total = (long)BN * Cout;
    if (t >= total) return;
    int i = (int)(t / Cout), c = (int)(t % Cout);
    int b = i / NPTS;
    const int* nb = idx + (size_t)i * KNN;
    int S = 2 * Cout;
    float vmax = -3.0e38f;
    for (int k = 0; k < KNN; ++k) {
        int j = b * NPTS + nb[k];
        vmax = fmaxf(vmax, UV[(size_t)j * S + Cout + c]);
    }
    Xout[(size_t)i * Cout + c] =
        UV[(size_t)i * S + c] - UV[(size_t)i * S + Cout + c] + bias[c] + vmax;
}

__global__ void prep_cat_kernel(const float* __restrict__ X1, const float* __restrict__ X2,
                                const float* __restrict__ X3, const float* __restrict__ X4,
                                _Float16* __restrict__ catH) {
    long t = blockIdx.x * (long)blockDim.x + threadIdx.x;
    if (t >= (long)BN * 512) return;
    int i = (int)(t / 512), c = (int)(t % 512);
    float v;
    if (c < 64)       v = X1[(size_t)i * 64 + c];
    else if (c < 128) v = X2[(size_t)i * 64 + (c - 64)];
    else if (c < 256) v = X3[(size_t)i * 128 + (c - 128)];
    else              v = X4[(size_t)i * 256 + (c - 256)];
    catH[t] = (_Float16)v;
}

// ---------------------------------------------------------------------------
// Classifier head on pooled [8,1024]: two ReLU linears + logits + log_softmax
// ---------------------------------------------------------------------------
__global__ void __launch_bounds__(512) head_kernel(const float* __restrict__ gmax,
                                                   const float* __restrict__ Wa, const float* __restrict__ ba,
                                                   const float* __restrict__ Wb, const float* __restrict__ bb,
                                                   const float* __restrict__ Wc, const float* __restrict__ bc,
                                                   float* __restrict__ out) {
    __shared__ float fin[1024];
    __shared__ float h1[512];
    __shared__ float h2[256];
    __shared__ float lg[23];
    int b = blockIdx.x, tid = threadIdx.x;
    fin[tid]       = gmax[(size_t)b * 1024 + tid];
    fin[tid + 512] = gmax[(size_t)b * 1024 + tid + 512];
    __syncthreads();
    {
        float s = ba[tid];
        for (int k = 0; k < 1024; ++k) s += fin[k] * Wa[(size_t)k * 512 + tid];
        h1[tid] = fmaxf(s, 0.f);
    }
    __syncthreads();
    if (tid < 256) {
        float s = bb[tid];
        for (int k = 0; k < 512; ++k) s += h1[k] * Wb[(size_t)k * 256 + tid];
        h2[tid] = fmaxf(s, 0.f);
    }
    __syncthreads();
    if (tid < 23) {
        float s = bc[tid];
        for (int k = 0; k < 256; ++k) s += h2[k] * Wc[(size_t)k * 23 + tid];
        lg[tid] = s;
    }
    __syncthreads();
    if (tid == 0) {
        float mx = lg[0];
        for (int c = 1; c < 23; ++c) mx = fmaxf(mx, lg[c]);
        float se = 0.f;
        for (int c = 0; c < 23; ++c) se += expf(lg[c] - mx);
        float l = logf(se);
        for (int c = 0; c < 23; ++c) out[b * 23 + c] = lg[c] - mx - l;
    }
}

// ---------------------------------------------------------------------------
extern "C" void kernel_launch(void* const* d_in, const int* in_sizes, int n_in,
                              void* d_out, int out_size, void* d_ws, size_t ws_size,
                              hipStream_t stream) {
    const float* pos = (const float*)d_in[0];
    const float* x   = (const float*)d_in[1];
    const float* W1  = (const float*)d_in[2];  const float* b1  = (const float*)d_in[3];
    const float* W2  = (const float*)d_in[4];  const float* b2  = (const float*)d_in[5];
    const float* W3  = (const float*)d_in[6];  const float* b3  = (const float*)d_in[7];
    const float* W4  = (const float*)d_in[8];  const float* b4  = (const float*)d_in[9];
    const float* Wf1 = (const float*)d_in[10]; const float* bf1 = (const float*)d_in[11];
    const float* Wa  = (const float*)d_in[12]; const float* ba  = (const float*)d_in[13];
    const float* Wb  = (const float*)d_in[14]; const float* bb  = (const float*)d_in[15];
    const float* Wc  = (const float*)d_in[16]; const float* bc  = (const float*)d_in[17];
    float* out = (float*)d_out;
    (void)in_sizes; (void)n_in; (void)out_size; (void)ws_size;

    char*  ws  = (char*)d_ws;
    size_t off = 0;
    auto alloc = [&](size_t bytes) -> void* {
        void* p = ws + off;
        off += (bytes + 255) & ~(size_t)255;
        return p;
    };
    _Float16* Xh   = (_Float16*)alloc((size_t)BN * 256 * sizeof(_Float16));
    float*    sqn  = (float*)   alloc((size_t)BN * sizeof(float));
    int*      idx  = (int*)     alloc((size_t)BN * KNN * sizeof(int));
    _Float16* Wcat = (_Float16*)alloc((size_t)128 * 512 * sizeof(_Float16));
    float*    UV   = (float*)   alloc((size_t)BN * 512 * sizeof(float));
    float*    X1   = (float*)   alloc((size_t)BN * 64 * sizeof(float));
    float*    X2   = (float*)   alloc((size_t)BN * 64 * sizeof(float));
    float*    X3   = (float*)   alloc((size_t)BN * 128 * sizeof(float));
    float*    X4   = (float*)   alloc((size_t)BN * 256 * sizeof(float));
    _Float16* catH = (_Float16*)alloc((size_t)BN * 512 * sizeof(_Float16));
    _Float16* Wf1h = (_Float16*)alloc((size_t)512 * 1024 * sizeof(_Float16));
    float*    gmax = (float*)   alloc((size_t)BATCH * 1024 * sizeof(float));

    dim3 kg(NPTS / 16, BATCH);
    const int PB = (BN + 255) / 256;

    // Layer 1: Cin=6(pad 32), Cout=64
    prep0_kernel<<<PB, 256, 0, stream>>>(pos, x, Xh, sqn);
    knn_kernel<1><<<kg, 32, 0, stream>>>(Xh, sqn, idx);
    prep_edge_w_kernel<<<(32 * 128 + 255) / 256, 256, 0, stream>>>(W1, Wcat, 6, 64, 32);
    gemm_kernel<<<dim3(128 / 32, BN / 64), 128, 0, stream>>>(Xh, Wcat, nullptr, UV, BN, 128, 32, 0);
    edge_agg_kernel<<<(BN * 64 + 255) / 256, 256, 0, stream>>>(UV, idx, b1, X1, 64);

    // Layer 2: Cin=64, Cout=64
    prep_feats_kernel<<<PB, 256, 0, stream>>>(X1, Xh, sqn, 64, 64);
    knn_kernel<2><<<kg, 32, 0, stream>>>(Xh, sqn, idx);
    prep_edge_w_kernel<<<(64 * 128 + 255) / 256, 256, 0, stream>>>(W2, Wcat, 64, 64, 64);
    gemm_kernel<<<dim3(128 / 32, BN / 64), 128, 0, stream>>>(Xh, Wcat, nullptr, UV, BN, 128, 64, 0);
    edge_agg_kernel<<<(BN * 64 + 255) / 256, 256, 0, stream>>>(UV, idx, b2, X2, 64);

    // Layer 3: Cin=64, Cout=128
    prep_feats_kernel<<<PB, 256, 0, stream>>>(X2, Xh, sqn, 64, 64);
    knn_kernel<2><<<kg, 32, 0, stream>>>(Xh, sqn, idx);
    prep_edge_w_kernel<<<(64 * 256 + 255) / 256, 256, 0, stream>>>(W3, Wcat, 64, 128, 64);
    gemm_kernel<<<dim3(256 / 32, BN / 64), 128, 0, stream>>>(Xh, Wcat, nullptr, UV, BN, 256, 64, 0);
    edge_agg_kernel<<<(BN * 128 + 255) / 256, 256, 0, stream>>>(UV, idx, b3, X3, 128);

    // Layer 4: Cin=128, Cout=256
    prep_feats_kernel<<<PB, 256, 0, stream>>>(X3, Xh, sqn, 128, 128);
    knn_kernel<4><<<kg, 32, 0, stream>>>(Xh, sqn, idx);
    prep_edge_w_kernel<<<(128 * 512 + 255) / 256, 256, 0, stream>>>(W4, Wcat, 128, 256, 128);
    gemm_kernel<<<dim3(512 / 32, BN / 64), 128, 0, stream>>>(Xh, Wcat, nullptr, UV, BN, 512, 128, 0);
    edge_agg_kernel<<<(BN * 256 + 255) / 256, 256, 0, stream>>>(UV, idx, b4, X4, 256);

    // concat -> Linear(512,1024) with fused max-over-N pooling
    prep_cat_kernel<<<(BN * 512 + 255) / 256, 256, 0, stream>>>(X1, X2, X3, X4, catH);
    conv_w_kernel<<<(512 * 1024 + 255) / 256, 256, 0, stream>>>(Wf1, Wf1h, 512 * 1024);
    fill_kernel<<<(BATCH * 1024 + 255) / 256, 256, 0, stream>>>(gmax, BATCH * 1024, -3.0e38f);
    gemm_kernel<<<dim3(1024 / 32, BN / 64), 128, 0, stream>>>(catH, Wf1h, bf1, gmax, BN, 1024, 512, NPTS);

    // classifier head + log_softmax
    head_kernel<<<BATCH, 512, 0, stream>>>(gmax, Wa, ba, Wb, bb, Wc, bc, out);
}
